// AttentiveReader_50070728736966
// MI455X (gfx1250) — matrix operand order repaired and checked by reading, not compiled, not run on hardware
//
#include <hip/hip_runtime.h>
#include <hip/hip_bf16.h>
#include <math.h>

// ---------------------------------------------------------------------------
// AttentiveReader for MI455X (gfx1250, wave32, WMMA bf16 16x16x32)
// ---------------------------------------------------------------------------
typedef __bf16 bf16;
typedef bf16  v16bf __attribute__((ext_vector_type(16)));
typedef float v8f   __attribute__((ext_vector_type(8)));

#define B_   32
#define S_   1024
#define Q_   64
#define E_   128
#define H_   256
#define G_   768      // 3*H
#define D2_  512      // 2*H
#define ENT_ 600
#define HPAD  264     // fp32 h row stride   (256 + 8)  -> 8-bank rotation
#define HBPAD 272     // bf16 h row stride   (256 + 16) -> 8-bank rotation
#define GPAD  776     // fp32 gh row stride  (768 + 8)

// ---- WMMA fragment helpers (contiguous k = 16*half + i in A and B, wave32) ----------
// A: 16x32 bf16 (MxK). lane<16: row=lane, elems 0..15 -> K 0..15
//                      lane>=16: row=lane-16, K 16..31.
// Two aligned 16B chunks per lane -> b128 loads (global or ds).
__device__ __forceinline__ v16bf load_a_bf(const bf16* A, int lda, int m_base,
                                           int k_base, int lane) {
  int half = (lane >> 4) & 1, m = lane & 15;
  const bf16* p = A + (size_t)(m_base + m) * lda + k_base + half * 16;
  v16bf a;
#pragma unroll
  for (int i = 0; i < 8; ++i) a[i] = p[i];
#pragma unroll
  for (int i = 0; i < 8; ++i) a[8 + i] = p[8 + i];
  return a;
}

// B: 32x16 bf16 (KxN) built from row-major W[N][K]: B[k][n] = W[n_base+n][k].
// lane<16: col n=lane, K = k_base+0..15 ; lane>=16: col n=lane-16, K = +16..31.
// Each lane reads 32 contiguous bytes -> two global_load_b128.
__device__ __forceinline__ v16bf load_b_wT(const bf16* W, int ldw, int n_base,
                                           int k_base, int lane) {
  int half = (lane >> 4) & 1, n = lane & 15;
  const bf16* p = W + (size_t)(n_base + n) * ldw + k_base + half * 16;
  v16bf b;
#pragma unroll
  for (int i = 0; i < 16; ++i) b[i] = p[i];
  return b;
}

__device__ __forceinline__ v8f wmma_bf16(v16bf a, v16bf b, v8f c) {
  return __builtin_amdgcn_wmma_f32_16x16x32_bf16(false, a, false, b,
                                                 (short)0, c, false, false);
}

// ---- small prep kernels ----------------------------------------------------
__global__ void cvt_bf16_k(const float* __restrict__ src, bf16* __restrict__ dst,
                           int n) {
  int i = blockIdx.x * blockDim.x + threadIdx.x;
  if (i < n) dst[i] = (bf16)src[i];
}

__global__ void gather_emb_k(const int* __restrict__ tok,
                             const float* __restrict__ emb,
                             bf16* __restrict__ A, int n) {
  int idx = blockIdx.x * blockDim.x + threadIdx.x;
  if (idx >= n) return;
  int ti = idx >> 7;           // token index
  int e  = idx & (E_ - 1);
  A[idx] = (bf16)emb[(size_t)tok[ti] * E_ + e];
}

// ---- xg = A(bf16)[Mtot x 128] @ Wih^T + bih  -> bf16 [2][Mtot][768] --------
// block = 256 (8 waves); each wave one 16x16 tile; grid (Mtot/16, 6, 2 dirs)
__global__ __launch_bounds__(256) void gates_gemm_k(
    const bf16* __restrict__ A, const bf16* __restrict__ Wih,
    const float* __restrict__ bih, bf16* __restrict__ xg, int Mtot) {
  const int dir = blockIdx.z;
  const bf16*  W  = Wih + (size_t)dir * G_ * E_;
  const float* bb = bih + dir * G_;
  bf16* xo = xg + (size_t)dir * Mtot * G_;

  const int w = threadIdx.x >> 5, lane = threadIdx.x & 31;
  const int mtile = blockIdx.x;
  const int ntile = blockIdx.y * 8 + w;

  v8f c = {0.f, 0.f, 0.f, 0.f, 0.f, 0.f, 0.f, 0.f};
#pragma unroll
  for (int kt = 0; kt < 4; ++kt) {   // K = 128 = 4 * 32
    v16bf a  = load_a_bf(A, E_, mtile * 16, kt * 32, lane);
    v16bf bf = load_b_wT(W, E_, ntile * 16, kt * 32, lane);
    c = wmma_bf16(a, bf, c);
  }
  const int n    = ntile * 16 + (lane & 15);
  const int half = (lane >> 4) & 1;
  const float bn = bb[n];
#pragma unroll
  for (int j = 0; j < 8; ++j) {
    int m = mtile * 16 + j + half * 8;
    xo[(size_t)m * G_ + n] = (bf16)(c[j] + bn);
  }
}

// ---- persistent bidirectional GRU: 4 workgroups = (net, dir) ---------------
// Per step: gh = h @ Whh^T with WMMA (A-frags cached in regs, bf16 h mirror
// in LDS), then fused gate math; next step's xg prefetched during the GEMM.
__global__ __launch_bounds__(1024) void gru_seq_k(
    const bf16* __restrict__ xgS, const bf16* __restrict__ xgQ,
    const bf16* __restrict__ WhhS, const bf16* __restrict__ WhhQ,
    const float* __restrict__ bhhS, const float* __restrict__ bhhQ,
    const int* __restrict__ lenS, const int* __restrict__ lenQ,
    float* __restrict__ y, float* __restrict__ qhn) {
  const int net = blockIdx.x >> 1;   // 0 = stories, 1 = questions
  const int dir = blockIdx.x & 1;    // 0 = fwd, 1 = bwd
  const int T   = net ? Q_ : S_;
  const bf16*  xg  = (net ? xgQ : xgS) + (size_t)dir * (size_t)(B_ * T) * G_;
  const bf16*  Whh = (net ? WhhQ : WhhS) + (size_t)dir * G_ * H_;
  const float* bhh = (net ? bhhQ : bhhS) + dir * G_;
  const int*   len = net ? lenQ : lenS;

  __shared__ __align__(16) float h_lds[B_][HPAD];   // fp32 master hidden
  __shared__ __align__(16) bf16  h_bf[B_][HBPAD];   // bf16 mirror for WMMA A
  __shared__ __align__(16) float gh_lds[B_][GPAD];  // per-step gh staging

  for (int i = threadIdx.x; i < B_ * HPAD; i += 1024) (&h_lds[0][0])[i] = 0.0f;
  for (int i = threadIdx.x; i < B_ * HBPAD; i += 1024)
    (&h_bf[0][0])[i] = (bf16)0.0f;
  __syncthreads();

  const int w     = threadIdx.x >> 5;
  const int lane  = threadIdx.x & 31;
  const int half  = (lane >> 4) & 1;
  const int nl    = lane & 15;
  const int mtile = w >> 4;          // waves 0-15 -> M rows 0-15, 16-31 -> 16-31
  const int nbase = (w & 15) * 3;    // 3 consecutive N-tiles per wave

  for (int step = 0; step < T; ++step) {
    const int t = dir ? (T - 1 - step) : step;

    // Prefetch next step's input-gate rows into cache (global_prefetch_b8),
    // overlapped with this step's WMMA work. 32 rows x 1536B = 384 lines.
    if (step + 1 < T && threadIdx.x < 384) {
      int tn = dir ? (T - 2 - step) : (step + 1);
      int pb = threadIdx.x / 12;        // batch
      int ch = threadIdx.x % 12;        // 128B chunk within row
      __builtin_prefetch(
          (const void*)(xg + ((size_t)pb * T + tn) * G_ + ch * 64), 0, 1);
    }

    // --- gh = h @ Whh^T : M=32, N=768, K=256 ---------------------------
    // Load this wave's 8 A-fragments once; reuse across 3 N-tiles.
    v16bf afrag[8];
#pragma unroll
    for (int kt = 0; kt < 8; ++kt)
      afrag[kt] = load_a_bf(&h_bf[0][0], HBPAD, mtile * 16, kt * 32, lane);

    for (int rr = 0; rr < 3; ++rr) {
      int ntile = nbase + rr;          // 0..47
      v8f c = {0.f, 0.f, 0.f, 0.f, 0.f, 0.f, 0.f, 0.f};
#pragma unroll
      for (int kt = 0; kt < 8; ++kt) {   // K = 256 = 8 * 32
        v16bf bf = load_b_wT(Whh, H_, ntile * 16, kt * 32, lane);
        c = wmma_bf16(afrag[kt], bf, c);
      }
      int n = ntile * 16 + nl;
#pragma unroll
      for (int j = 0; j < 8; ++j)
        gh_lds[mtile * 16 + j + half * 8][n] = c[j];
    }
    __syncthreads();

    // --- fused GRU gate update: 32*256 states over 1024 threads ----------
#pragma unroll
    for (int kk = 0; kk < 8; ++kk) {
      int idx = threadIdx.x + kk * 1024;
      int b = idx >> 8;
      int j = idx & (H_ - 1);
      float hr = gh_lds[b][j]          + bhh[j];
      float hz = gh_lds[b][j + H_]     + bhh[j + H_];
      float hn = gh_lds[b][j + 2*H_]   + bhh[j + 2*H_];
      size_t row = ((size_t)b * T + t) * G_;
      float xr = (float)xg[row + j];
      float xz = (float)xg[row + H_ + j];
      float xn = (float)xg[row + 2*H_ + j];
      float rg = 1.0f / (1.0f + expf(-(xr + hr)));
      float zg = 1.0f / (1.0f + expf(-(xz + hz)));
      float ng = tanhf(xn + rg * hn);
      float hp = h_lds[b][j];
      float hv = (1.0f - zg) * ng + zg * hp;
      bool valid = t < len[b];
      float ho = valid ? hv : hp;
      h_lds[b][j] = ho;
      h_bf[b][j]  = (bf16)ho;
      if (net == 0)
        y[((size_t)b * S_ + t) * D2_ + dir * H_ + j] = valid ? ho : 0.0f;
    }
    __syncthreads();
  }

  if (net == 1) {  // final hidden for questions net
#pragma unroll
    for (int kk = 0; kk < 8; ++kk) {
      int idx = threadIdx.x + kk * 1024;
      int b = idx >> 8;
      int j = idx & (H_ - 1);
      qhn[b * D2_ + dir * H_ + j] = h_lds[b][j];
    }
  }
}

// ---- attention / readout ---------------------------------------------------
__global__ void qmix_k(const float* __restrict__ mix,
                       const float* __restrict__ qhn, float* __restrict__ u) {
  int b = blockIdx.x, d = threadIdx.x;
  const float* qb = qhn + b * D2_;
  float acc = 0.f;
  for (int e = 0; e < D2_; ++e) acc += mix[(size_t)d * D2_ + e] * qb[e];
  u[b * D2_ + d] = acc;
}

__global__ __launch_bounds__(1024) void attn_scores_k(
    const float* __restrict__ y, const float* __restrict__ u,
    const int* __restrict__ stories, float* __restrict__ wout) {
  int b = blockIdx.x, s = threadIdx.x;
  __shared__ float sm[S_];
  __shared__ float red[S_];
  const float* yr = y + ((size_t)b * S_ + s) * D2_;
  const float* ub = u + b * D2_;
  float acc = 0.f;
  for (int d = 0; d < D2_; ++d) acc += yr[d] * ub[d];
  if (stories[b * S_ + s] <= 0) acc -= 100.0f;
  sm[s] = acc; red[s] = acc;
  __syncthreads();
  for (int o = 512; o > 0; o >>= 1) {
    if (s < o) red[s] = fmaxf(red[s], red[s + o]);
    __syncthreads();
  }
  float mx = red[0];
  __syncthreads();
  float e = expf(sm[s] - mx);
  red[s] = e;
  __syncthreads();
  for (int o = 512; o > 0; o >>= 1) {
    if (s < o) red[s] += red[s + o];
    __syncthreads();
  }
  wout[b * S_ + s] = e / red[0];
}

__global__ void attn_reduce_k(const float* __restrict__ y,
                              const float* __restrict__ w,
                              float* __restrict__ r) {
  int b = blockIdx.x;
  int d = blockIdx.y * 256 + threadIdx.x;
  const float* wb = w + b * S_;
  const float* yb = y + (size_t)b * S_ * D2_ + d;
  float acc = 0.f;
  for (int s = 0; s < S_; ++s) acc += yb[(size_t)s * D2_] * wb[s];
  r[b * D2_ + d] = acc;
}

__global__ void out_k(const float* __restrict__ r, const float* __restrict__ W,
                      const float* __restrict__ bias, float* __restrict__ out) {
  int b = blockIdx.x, e = threadIdx.x;
  if (e >= ENT_) return;
  const float* rb = r + b * D2_;
  const float* wr = W + (size_t)e * D2_;
  float acc = bias[e];
  for (int k = 0; k < D2_; ++k) acc += rb[k] * wr[k];
  out[b * ENT_ + e] = acc;
}

// ---------------------------------------------------------------------------
extern "C" void kernel_launch(void* const* d_in, const int* in_sizes, int n_in,
                              void* d_out, int out_size, void* d_ws,
                              size_t ws_size, hipStream_t stream) {
  const int*   stories   = (const int*)d_in[0];
  const int*   questions = (const int*)d_in[1];
  const int*   s_len = (const int*)d_in[2];
  const int*   q_len = (const int*)d_in[3];
  const float* emb   = (const float*)d_in[4];
  const float* s_Wih = (const float*)d_in[5];
  const float* s_Whh = (const float*)d_in[6];
  const float* s_bih = (const float*)d_in[7];
  const float* s_bhh = (const float*)d_in[8];
  const float* q_Wih = (const float*)d_in[9];
  const float* q_Whh = (const float*)d_in[10];
  const float* q_bih = (const float*)d_in[11];
  const float* q_bhh = (const float*)d_in[12];
  const float* mix   = (const float*)d_in[13];
  const float* out_W = (const float*)d_in[14];
  const float* out_b = (const float*)d_in[15];
  float* out = (float*)d_out;

  char* ws = (char*)d_ws;
  size_t off = 0;
  auto take = [&](size_t bytes) -> char* {
    char* p = ws + off;
    off = (off + bytes + 255) & ~(size_t)255;
    return p;
  };
  bf16* WihS = (bf16*)take((size_t)2 * G_ * E_ * sizeof(bf16));
  bf16* WihQ = (bf16*)take((size_t)2 * G_ * E_ * sizeof(bf16));
  bf16* WhhS = (bf16*)take((size_t)2 * G_ * H_ * sizeof(bf16));
  bf16* WhhQ = (bf16*)take((size_t)2 * G_ * H_ * sizeof(bf16));
  bf16* As   = (bf16*)take((size_t)B_ * S_ * E_ * sizeof(bf16));
  bf16* Aq   = (bf16*)take((size_t)B_ * Q_ * E_ * sizeof(bf16));
  bf16* xgS  = (bf16*)take((size_t)2 * B_ * S_ * G_ * sizeof(bf16));
  bf16* xgQ  = (bf16*)take((size_t)2 * B_ * Q_ * G_ * sizeof(bf16));
  float* y   = (float*)take((size_t)B_ * S_ * D2_ * sizeof(float));
  float* qhn = (float*)take((size_t)B_ * D2_ * sizeof(float));
  float* u   = (float*)take((size_t)B_ * D2_ * sizeof(float));
  float* aw  = (float*)take((size_t)B_ * S_ * sizeof(float));
  float* r   = (float*)take((size_t)B_ * D2_ * sizeof(float));
  (void)ws_size; (void)in_sizes; (void)n_in; (void)out_size;

  // 1) weights -> bf16
  {
    int n1 = 2 * G_ * E_, n2 = 2 * G_ * H_;
    cvt_bf16_k<<<(n1 + 255) / 256, 256, 0, stream>>>(s_Wih, WihS, n1);
    cvt_bf16_k<<<(n1 + 255) / 256, 256, 0, stream>>>(q_Wih, WihQ, n1);
    cvt_bf16_k<<<(n2 + 255) / 256, 256, 0, stream>>>(s_Whh, WhhS, n2);
    cvt_bf16_k<<<(n2 + 255) / 256, 256, 0, stream>>>(q_Whh, WhhQ, n2);
  }
  // 2) embedding gather -> bf16
  {
    int ns = B_ * S_ * E_, nq = B_ * Q_ * E_;
    gather_emb_k<<<(ns + 255) / 256, 256, 0, stream>>>(stories, emb, As, ns);
    gather_emb_k<<<(nq + 255) / 256, 256, 0, stream>>>(questions, emb, Aq, nq);
  }
  // 3) xg = A @ Wih^T + bih (WMMA bf16)
  {
    dim3 gs(B_ * S_ / 16, G_ / (16 * 8), 2);   // (2048, 6, 2)
    dim3 gq(B_ * Q_ / 16, G_ / (16 * 8), 2);   // (128, 6, 2)
    gates_gemm_k<<<gs, 256, 0, stream>>>(As, WihS, s_bih, xgS, B_ * S_);
    gates_gemm_k<<<gq, 256, 0, stream>>>(Aq, WihQ, q_bih, xgQ, B_ * Q_);
  }
  // 4) persistent bidirectional GRUs (4 workgroups, 32 waves each)
  gru_seq_k<<<4, 1024, 0, stream>>>(xgS, xgQ, WhhS, WhhQ, s_bhh, q_bhh,
                                    s_len, q_len, y, qhn);
  // 5) attention + readout
  qmix_k<<<B_, D2_, 0, stream>>>(mix, qhn, u);
  attn_scores_k<<<B_, S_, 0, stream>>>(y, u, stories, aw);
  attn_reduce_k<<<dim3(B_, 2), 256, 0, stream>>>(y, aw, r);
  out_k<<<B_, 640, 0, stream>>>(r, out_W, out_b, out);
}
